// LSTM_Encoder_46437186404872
// MI455X (gfx1250) — matrix operand order repaired and checked by reading, not compiled, not run on hardware
//
#include <hip/hip_runtime.h>

typedef __attribute__((ext_vector_type(16))) _Float16 v16h;
typedef __attribute__((ext_vector_type(8)))  _Float16 v8h;
typedef __attribute__((ext_vector_type(8)))  float    v8f;
typedef __attribute__((ext_vector_type(4)))  float    f32x4;

#define BATCH 64
#define SEQ   80
#define WLEN  8
#define NW    (BATCH*SEQ)   // 5120 words / rows per timestep-major matrix
#define KC    64            // char input K (50 -> 64)
#define NCG   416           // char gate cols: 2 dirs * 208 (200 -> 208)
#define KP    416           // main input K (400 -> 416)
#define KH    224           // recurrent K (200 -> 224)
#define NG    1664          // main gate cols: 2 dirs * 832 (13 hgroups * 4 gates * 16)
#define NPAIR 3081          // triu(79, k=1)

union V16H { v16h v; v8h h[2]; };

__device__ __forceinline__ v8f v8f_zero() {
  v8f z;
#pragma unroll
  for (int i = 0; i < 8; ++i) z[i] = 0.f;
  return z;
}
__device__ __forceinline__ float sigmoidf_(float x) { return 1.f / (1.f + __expf(-x)); }

// ---------------------------------------------------------------------------
// Generic WMMA GEMM: C[M][N] = A[M][Kp](f16,row-major) * Bt[N][Kp](f16)^T
// One wave per 16x32 output tile. Kp % 32 == 0, N % 32 == 0, M % 16 == 0.
// Per the CDNA5 16-bit A layout: lane holds K chunks [kb,kb+8) and [kb+16,kb+24),
// kb = (lane>>4)*8. B fragment: same K chunks [kb,kb+8) and [kb+16,kb+24).
// ---------------------------------------------------------------------------
__global__ __launch_bounds__(32) void gemm16(const _Float16* __restrict__ A,
                                             const _Float16* __restrict__ Bt,
                                             float* __restrict__ C, int Kp, int N) {
  const int lane = threadIdx.x & 31;
  const int m0 = blockIdx.x * 16;
  const int n0 = blockIdx.y * 32;
  v8f acc0 = v8f_zero(), acc1 = v8f_zero();
  const _Float16* arow = A + (size_t)(m0 + (lane & 15)) * Kp + ((lane >> 4) << 3);
  const _Float16* brow0 = Bt + (size_t)(n0 + (lane & 15)) * Kp + ((lane >> 4) << 3);
  const _Float16* brow1 = brow0 + (size_t)16 * Kp;
  for (int k0 = 0; k0 < Kp; k0 += 32) {
    V16H a, b0, b1;
    a.h[0]  = *(const v8h*)(arow + k0);
    a.h[1]  = *(const v8h*)(arow + k0 + 16);
    b0.h[0] = *(const v8h*)(brow0 + k0);
    b0.h[1] = *(const v8h*)(brow0 + k0 + 16);
    b1.h[0] = *(const v8h*)(brow1 + k0);
    b1.h[1] = *(const v8h*)(brow1 + k0 + 16);
    acc0 = __builtin_amdgcn_wmma_f32_16x16x32_f16(false, a.v, false, b0.v, (short)0, acc0, false, false);
    acc1 = __builtin_amdgcn_wmma_f32_16x16x32_f16(false, a.v, false, b1.v, (short)0, acc1, false, false);
  }
  const int cm = m0 + ((lane >> 4) << 3);
  const int cn = lane & 15;
#pragma unroll
  for (int v = 0; v < 8; ++v) {
    C[(size_t)(cm + v) * N + n0 + cn]      = acc0[v];
    C[(size_t)(cm + v) * N + n0 + 16 + cn] = acc1[v];
  }
}

// ---------------------------------------------------------------------------
// Weight prep: f32 -> f16, K padding, and gate-row permutation for the main
// LSTM so each 16-wide hidden group's i/f/g/o tiles are 16 columns apart.
// Also zeroes layer-1 input padding columns.
// ---------------------------------------------------------------------------
struct WPtrs {
  const float* c_wih[2]; const float* c_whh[2]; const float* c_bih[2]; const float* c_bhh[2];
  const float* wih[2][2]; const float* whh[2][2]; const float* bih[2][2]; const float* bhh[2][2];
};

__global__ __launch_bounds__(256) void prep_weights(WPtrs P,
    _Float16* __restrict__ cwih, float* __restrict__ cbias,
    _Float16* __restrict__ wihp, _Float16* __restrict__ whhp,
    float* __restrict__ biasp, _Float16* __restrict__ x1) {
  const int S0 = NCG * KC;
  const int S1 = NCG;
  const int S2 = 2 * NG * KP;
  const int S3 = 2 * NG * KH;
  const int S4 = 2 * NG;
  const int S5 = SEQ * BATCH * 16;
  const int total = S0 + S1 + S2 + S3 + S4 + S5;
  for (int i = blockIdx.x * blockDim.x + threadIdx.x; i < total; i += gridDim.x * blockDim.x) {
    int e = i;
    if (e < S0) {
      int n = e / KC, k = e % KC;
      int dir = n / 208, r = n % 208;
      float v = (r < 200 && k < 50) ? P.c_wih[dir][r * 50 + k] : 0.f;
      cwih[e] = (_Float16)v;
    } else if ((e -= S0) < S1) {
      int dir = e / 208, r = e % 208;
      cbias[e] = (r < 200) ? (P.c_bih[dir][r] + P.c_bhh[dir][r]) : 0.f;
    } else if ((e -= S1) < S2) {
      int l = e / (NG * KP), rem = e % (NG * KP);
      int n = rem / KP, k = rem % KP;
      int dir = n / 832, n2 = n % 832;
      int hg = n2 >> 6, g = (n2 >> 4) & 3, j = hg * 16 + (n2 & 15);
      float v = (j < 200 && k < 400) ? P.wih[l][dir][(size_t)(g * 200 + j) * 400 + k] : 0.f;
      wihp[e] = (_Float16)v;
    } else if ((e -= S2) < S3) {
      int l = e / (NG * KH), rem = e % (NG * KH);
      int n = rem / KH, k = rem % KH;
      int dir = n / 832, n2 = n % 832;
      int hg = n2 >> 6, g = (n2 >> 4) & 3, j = hg * 16 + (n2 & 15);
      float v = (j < 200 && k < 200) ? P.whh[l][dir][(size_t)(g * 200 + j) * 200 + k] : 0.f;
      whhp[e] = (_Float16)v;
    } else if ((e -= S3) < S4) {
      int l = e / NG, n = e % NG;
      int dir = n / 832, n2 = n % 832;
      int hg = n2 >> 6, g = (n2 >> 4) & 3, j = hg * 16 + (n2 & 15);
      biasp[e] = (j < 200) ? (P.bih[l][dir][g * 200 + j] + P.bhh[l][dir][g * 200 + j]) : 0.f;
    } else {
      e -= S4;
      int row = e / 16, c = e % 16;
      x1[(size_t)row * KP + 400 + c] = (_Float16)0.f;
    }
  }
}

// ---------------------------------------------------------------------------
// Embedding gathers: word part of x0 [S][B][KP] (f16) and char embeddings
// ce [NW*WLEN][KC] (f16, K-padded).
// ---------------------------------------------------------------------------
__global__ __launch_bounds__(256) void embed_gather(
    const int* __restrict__ widx, const int* __restrict__ cidx,
    const float* __restrict__ ext, const float* __restrict__ wemb,
    const float* __restrict__ cemb, _Float16* __restrict__ x0, _Float16* __restrict__ ce) {
  const int T0 = SEQ * BATCH * KP;
  const int T1 = NW * WLEN * KC;
  for (int i = blockIdx.x * blockDim.x + threadIdx.x; i < T0 + T1; i += gridDim.x * blockDim.x) {
    if (i < T0) {
      int col = i % KP, r = i / KP;
      int b = r % BATCH, s = r / BATCH;
      float v = 0.f;
      if (col < 300) {
        int w = widx[b * SEQ + s];
        int win = (w >= 50000) ? 1 : w;   // OOV -> UNK
        v = ext[(size_t)w * 300 + col] + wemb[(size_t)win * 300 + col];
      }
      x0[i] = (_Float16)v;
    } else {
      int e = i - T0;
      int k = e % KC, r = e / KC;        // r = word*8 + t
      float v = (k < 50) ? cemb[(size_t)cidx[r] * 50 + k] : 0.f;
      ce[(size_t)r * KC + k] = (_Float16)v;
    }
  }
}

// ---------------------------------------------------------------------------
// Char LSTM recurrence (hid=50, 8 steps). Block = 4 words x 64 threads.
// Thread owns hidden unit j of one word; h shared via LDS.
// ---------------------------------------------------------------------------
__global__ __launch_bounds__(256) void char_rec(
    const float* __restrict__ cg, const float* __restrict__ whhF,
    const float* __restrict__ whhB, const float* __restrict__ cbias,
    _Float16* __restrict__ x0) {
  const int dir = blockIdx.y;
  const int slot = threadIdx.x >> 6;
  const int j = threadIdx.x & 63;
  const int w = blockIdx.x * 4 + slot;
  __shared__ float hsh[4][52];
  const float* whh = dir ? whhB : whhF;
  const bool act = (j < 50);
  float c = 0.f, h = 0.f;
  if (act) hsh[slot][j] = 0.f;
  __syncthreads();
  for (int tt = 0; tt < WLEN; ++tt) {
    int t = dir ? (WLEN - 1 - tt) : tt;
    float g4[4];
    if (act) {
      size_t row = (size_t)(w * WLEN + t) * NCG + dir * 208;
#pragma unroll
      for (int g = 0; g < 4; ++g) {
        float acc = cg[row + g * 50 + j] + cbias[dir * 208 + g * 50 + j];
        const float* wr = whh + (size_t)(g * 50 + j) * 50;
        for (int k = 0; k < 50; ++k) acc += hsh[slot][k] * wr[k];
        g4[g] = acc;
      }
    }
    __syncthreads();
    if (act) {
      float si = sigmoidf_(g4[0]), sf = sigmoidf_(g4[1]);
      float cgn = tanhf(g4[2]),   so = sigmoidf_(g4[3]);
      c = sf * c + si * cgn;
      h = so * tanhf(c);
      hsh[slot][j] = h;
    }
    __syncthreads();
  }
  if (act) {
    int b = w / SEQ, s = w % SEQ;
    x0[((size_t)(s * BATCH + b)) * KP + 300 + dir * 50 + j] = (_Float16)h;
  }
}

// ---------------------------------------------------------------------------
// Fused main-LSTM recurrence: one 13-wave workgroup per direction runs the
// whole 80-step scan. h double-buffered in LDS (f16); wave hg owns hidden
// columns [hg*16, hg*16+16) and computes its i/f/g/o 16x16 tiles with WMMA;
// c lives in registers; one barrier per step.
// ---------------------------------------------------------------------------
__global__ __launch_bounds__(416) void lstm_rec(
    const float* __restrict__ gin, const _Float16* __restrict__ whhp,
    const float* __restrict__ bias, _Float16* __restrict__ xout,
    float* __restrict__ hout) {
  __shared__ _Float16 hbuf[2][BATCH][KH] __attribute__((aligned(16)));
  const int dir = blockIdx.x;
  const int lane = threadIdx.x & 31;
  const int hg = threadIdx.x >> 5;              // 0..12
  for (int i = threadIdx.x; i < 2 * BATCH * KH; i += blockDim.x)
    (&hbuf[0][0][0])[i] = (_Float16)0.f;
  __syncthreads();

  const int l15 = lane & 15;
  const int lhi = lane >> 4;
  const int jn = hg * 16 + l15;                 // hidden index this lane produces
  const size_t colbase = (size_t)dir * 832 + hg * 64;

  const float bI = bias[colbase + 0  + l15];
  const float bF = bias[colbase + 16 + l15];
  const float bG = bias[colbase + 32 + l15];
  const float bO = bias[colbase + 48 + l15];

  const _Float16* wrow[4];
#pragma unroll
  for (int g = 0; g < 4; ++g)
    wrow[g] = whhp + (colbase + g * 16 + l15) * KH + (lhi << 3);

  float c[4][8];
#pragma unroll
  for (int mt = 0; mt < 4; ++mt)
#pragma unroll
    for (int v = 0; v < 8; ++v) c[mt][v] = 0.f;

  int cur = 0;
  for (int tt = 0; tt < SEQ; ++tt) {
    const int t = dir ? (SEQ - 1 - tt) : tt;
#pragma unroll
    for (int mt = 0; mt < 4; ++mt) {
      v8f acc[4];
#pragma unroll
      for (int g = 0; g < 4; ++g) acc[g] = v8f_zero();
      const _Float16* arow = &hbuf[cur][mt * 16 + l15][lhi << 3];
#pragma unroll
      for (int kk = 0; kk < 7; ++kk) {
        V16H a;
        a.h[0] = *(const v8h*)(arow + kk * 32);
        a.h[1] = *(const v8h*)(arow + kk * 32 + 16);
#pragma unroll
        for (int g = 0; g < 4; ++g) {
          V16H b;
          b.h[0] = *(const v8h*)(wrow[g] + kk * 32);
          b.h[1] = *(const v8h*)(wrow[g] + kk * 32 + 16);
          acc[g] = __builtin_amdgcn_wmma_f32_16x16x32_f16(false, a.v, false, b.v, (short)0, acc[g], false, false);
        }
      }
      const int mb = mt * 16 + (lhi << 3);
#pragma unroll
      for (int v = 0; v < 8; ++v) {
        const int m = mb + v;
        const size_t grow = (size_t)(t * BATCH + m) * NG + colbase + l15;
        float gI = acc[0][v] + gin[grow + 0]  + bI;
        float gF = acc[1][v] + gin[grow + 16] + bF;
        float gG = acc[2][v] + gin[grow + 32] + bG;
        float gO = acc[3][v] + gin[grow + 48] + bO;
        float si = sigmoidf_(gI), sf = sigmoidf_(gF), so = sigmoidf_(gO);
        float cn = sf * c[mt][v] + si * tanhf(gG);
        c[mt][v] = cn;
        float hn = so * tanhf(cn);
        if (jn < 200) {
          hbuf[cur ^ 1][m][jn] = (_Float16)hn;
          const size_t orow = (size_t)(t * BATCH + m);
          if (xout) xout[orow * KP + dir * 200 + jn] = (_Float16)hn;
          if (hout) hout[orow * 400 + dir * 200 + jn] = hn;
        }
      }
    }
    __syncthreads();
    cur ^= 1;
  }
}

// ---------------------------------------------------------------------------
// Span features: out[b][pair][0:200] = fwd[q]-fwd[p], [200:400] = bwd[p+1]-bwd[q+1].
// Pure streaming: b128 loads/stores; 315 MB output => HBM-bound stage.
// ---------------------------------------------------------------------------
__global__ __launch_bounds__(256) void span_kernel(const float* __restrict__ h,
                                                   float* __restrict__ out) {
  const int pair = blockIdx.x;
  int p = 0, e = pair, rem = SEQ - 2;           // row p has (78 - p) pairs
  while (e >= rem) { e -= rem; --rem; ++p; }
  const int q = p + 1 + e;
  const float* hq  = h + (size_t)q * BATCH * 400;
  const float* hp  = h + (size_t)p * BATCH * 400;
  const float* hp1 = h + (size_t)(p + 1) * BATCH * 400;
  const float* hq1 = h + (size_t)(q + 1) * BATCH * 400;
  for (int i = threadIdx.x; i < BATCH * 100; i += blockDim.x) {
    const int b = i / 100, col = (i % 100) * 4;
    f32x4 v;
    if (col < 200)
      v = *(const f32x4*)(hq + b * 400 + col) - *(const f32x4*)(hp + b * 400 + col);
    else
      v = *(const f32x4*)(hp1 + b * 400 + col) - *(const f32x4*)(hq1 + b * 400 + col);
    *(f32x4*)(out + ((size_t)b * NPAIR + pair) * 400 + col) = v;
  }
}

// ---------------------------------------------------------------------------
extern "C" void kernel_launch(void* const* d_in, const int* in_sizes, int n_in,
                              void* d_out, int out_size, void* d_ws, size_t ws_size,
                              hipStream_t stream) {
  (void)in_sizes; (void)n_in; (void)out_size; (void)ws_size;
  const int*   widx = (const int*)d_in[0];
  const int*   cidx = (const int*)d_in[1];
  const float* ext  = (const float*)d_in[2];
  const float* wemb = (const float*)d_in[3];
  const float* cemb = (const float*)d_in[4];

  WPtrs P;
  P.c_wih[0] = (const float*)d_in[5];  P.c_whh[0] = (const float*)d_in[6];
  P.c_bih[0] = (const float*)d_in[7];  P.c_bhh[0] = (const float*)d_in[8];
  P.c_wih[1] = (const float*)d_in[9];  P.c_whh[1] = (const float*)d_in[10];
  P.c_bih[1] = (const float*)d_in[11]; P.c_bhh[1] = (const float*)d_in[12];
  for (int l = 0; l < 2; ++l) {
    const int base = 13 + 8 * l;
    P.wih[l][0] = (const float*)d_in[base + 0]; P.whh[l][0] = (const float*)d_in[base + 1];
    P.bih[l][0] = (const float*)d_in[base + 2]; P.bhh[l][0] = (const float*)d_in[base + 3];
    P.wih[l][1] = (const float*)d_in[base + 4]; P.whh[l][1] = (const float*)d_in[base + 5];
    P.bih[l][1] = (const float*)d_in[base + 6]; P.bhh[l][1] = (const float*)d_in[base + 7];
  }

  char* ws = (char*)d_ws;
  size_t off = 0;
  auto take = [&](size_t bytes) -> void* {
    void* p = ws + off;
    off = (off + bytes + 255) & ~(size_t)255;
    return p;
  };
  _Float16* x0    = (_Float16*)take((size_t)SEQ * BATCH * KP * 2);
  _Float16* x1    = (_Float16*)take((size_t)SEQ * BATCH * KP * 2);
  _Float16* ce    = (_Float16*)take((size_t)NW * WLEN * KC * 2);
  float*    cg    = (float*)   take((size_t)NW * WLEN * NCG * 4);
  _Float16* cwih  = (_Float16*)take((size_t)NCG * KC * 2);
  float*    cbias = (float*)   take((size_t)NCG * 4);
  _Float16* wihp  = (_Float16*)take((size_t)2 * NG * KP * 2);
  _Float16* whhp  = (_Float16*)take((size_t)2 * NG * KH * 2);
  float*    biasp = (float*)   take((size_t)2 * NG * 4);
  float*    gbuf  = (float*)   take((size_t)NW * NG * 4);
  float*    hout  = (float*)   take((size_t)SEQ * BATCH * 400 * 4);

  prep_weights<<<2048, 256, 0, stream>>>(P, cwih, cbias, wihp, whhp, biasp, x1);
  embed_gather<<<8192, 256, 0, stream>>>(widx, cidx, ext, wemb, cemb, x0, ce);

  // char input projection: [40960 x 64] x [64 x 416]
  gemm16<<<dim3(NW * WLEN / 16, NCG / 32), 32, 0, stream>>>(ce, cwih, cg, KC, NCG);
  char_rec<<<dim3(NW / 4, 2), 256, 0, stream>>>(cg, (const float*)d_in[6], (const float*)d_in[10], cbias, x0);

  // layer 0
  gemm16<<<dim3(NW / 16, NG / 32), 32, 0, stream>>>(x0, wihp, gbuf, KP, NG);
  lstm_rec<<<2, 416, 0, stream>>>(gbuf, whhp, biasp, x1, nullptr);

  // layer 1
  gemm16<<<dim3(NW / 16, NG / 32), 32, 0, stream>>>(x1, wihp + (size_t)NG * KP, gbuf, KP, NG);
  lstm_rec<<<2, 416, 0, stream>>>(gbuf, whhp + (size_t)NG * KH, biasp + NG, nullptr, hout);

  span_kernel<<<NPAIR, 256, 0, stream>>>(hout, (float*)d_out);
}
